// VarianceAdaptor_30846455120722
// MI455X (gfx1250) — matrix operand order repaired and checked by reading, not compiled, not run on hardware
//
#include <hip/hip_runtime.h>
#include <hip/hip_fp16.h>

typedef __attribute__((ext_vector_type(16))) _Float16 v16h;
typedef __attribute__((ext_vector_type(2)))  _Float16 h2;
typedef __attribute__((ext_vector_type(8)))  float    v8f;
typedef __attribute__((ext_vector_type(4)))  unsigned int u32x4;
typedef __attribute__((ext_vector_type(8)))  int      i32x8;
typedef __attribute__((ext_vector_type(4)))  int      i32x4;

#define HDIM   256
#define TSRC   512
#define MAXLEN 2048
#define BATCH  64
#define KIM    768      // 3 * 256 im2col K
#define NB     256      // bins+1 / embedding rows

// CDNA5 async global->LDS copy, 16B per lane, tracked by ASYNCcnt.
#define ASYNC_CP16(ldsoff, gaddr)                                          \
  asm volatile("global_load_async_to_lds_b128 %0, %1, off"                 \
               :: "v"((unsigned)(ldsoff)), "v"((unsigned long long)(gaddr)) \
               : "memory")
#define WAIT_ASYNC() asm volatile("s_wait_asynccnt 0x0" ::: "memory")
// wait with dummy register deps: pins all 8 WMMAs of the chunk before the stall
#define WAIT_ASYNC_DEP(a0,a1,a2,a3,a4,a5,a6,a7)                            \
  asm volatile("s_wait_asynccnt 0x0"                                       \
               :: "v"(a0),"v"(a1),"v"(a2),"v"(a3),"v"(a4),"v"(a5),"v"(a6),"v"(a7) \
               : "memory")

// ---------------------------------------------------------------------------
// TDM: 2D tile load Global->LDS. rows x 512B, LDS row stride 528B via D# pad.
// data_size=8B; tensor_dim0 = tile_dim0 = 64 units; tensor_dim1 = rows_avail
// (rows past it are hardware-zeroed); pad: 512B interval, 16B amount.
// ---------------------------------------------------------------------------
#if __has_builtin(__builtin_amdgcn_tensor_load_to_lds)
__device__ __forceinline__ void tdm_load_rows(unsigned lds_addr, const void* gptr,
                                              int rows, int rows_avail)
{
  unsigned long long ga = (unsigned long long)gptr;
  u32x4 g0;
  g0[0] = 1u;                                            // count=1, user mode
  g0[1] = lds_addr;                                      // lds_addr (bytes)
  g0[2] = (unsigned)(ga & 0xffffffffu);                  // global_addr[31:0]
  g0[3] = (unsigned)((ga >> 32) & 0x01ffffffu) | (2u << 30);  // addr[56:32] | type=2
  i32x8 g1;
  g1[0] = (3 << 16) | (1 << 20) | (6 << 22) | (3 << 25); // 8B elems, pad_en, 512B/16B
  g1[1] = (64 << 16);                                    // tensor_dim0 lo16 = 64
  g1[2] = (rows_avail & 0xffff) << 16;                   // dim0 hi=0 | tensor_dim1 lo16
  g1[3] = ((rows_avail >> 16) & 0xffff) | (64 << 16);    // tensor_dim1 hi | tile_dim0=64
  g1[4] = rows & 0xffff;                                 // tile_dim1 | tile_dim2=0
  g1[5] = 64;                                            // tensor_dim0_stride = 64 units
  g1[6] = 0;
  g1[7] = 0;
  #pragma unroll
  for (int k = 0; k < 4; ++k) g0[k] = (unsigned)__builtin_amdgcn_readfirstlane((int)g0[k]);
  #pragma unroll
  for (int k = 0; k < 8; ++k) g1[k] = __builtin_amdgcn_readfirstlane(g1[k]);
  i32x4 gz = {0, 0, 0, 0};
#if defined(__clang_major__) && __clang_major__ >= 23
  i32x8 gz8 = {0, 0, 0, 0, 0, 0, 0, 0};
  __builtin_amdgcn_tensor_load_to_lds(g0, g1, gz, gz, gz8, 0);
#else
  __builtin_amdgcn_tensor_load_to_lds(g0, g1, gz, gz, 0);
#endif
}
#endif

// ---------------------------------------------------------------------------
// 1) cumsum + searchsorted(right) -> gather idx, mel_len, mel_mask, dur_rounded
// ---------------------------------------------------------------------------
__global__ __launch_bounds__(TSRC)
void idx_kernel(const int* __restrict__ dur, int* __restrict__ idx,
                unsigned char* __restrict__ valid,
                float* __restrict__ out_durr, float* __restrict__ out_mellen,
                float* __restrict__ out_melmask)
{
  __shared__ int c[TSRC];
  const int b = blockIdx.x, t = threadIdx.x;
  const int d = dur[(size_t)b * TSRC + t];
  c[t] = d;
  __syncthreads();
  for (int off = 1; off < TSRC; off <<= 1) {
    int v = 0;
    if (t >= off) v = c[t - off];
    __syncthreads();
    c[t] += v;
    __syncthreads();
  }
  const int mel_len = c[TSRC - 1];
  if (t == 0) out_mellen[b] = (float)mel_len;
  out_durr[(size_t)b * TSRC + t] = (float)d;
  for (int f = t; f < MAXLEN; f += TSRC) {
    int lo = 0, hi = TSRC;                 // first i with c[i] > f (side='right')
    while (lo < hi) { int m = (lo + hi) >> 1; if (c[m] <= f) lo = m + 1; else hi = m; }
    int id = lo < TSRC ? lo : TSRC - 1;
    bool vld = f < mel_len;
    idx[(size_t)b * MAXLEN + f] = id;
    valid[(size_t)b * MAXLEN + f] = vld ? 1 : 0;
    out_melmask[(size_t)b * MAXLEN + f] = vld ? 0.f : 1.f;
  }
}

// ---------------------------------------------------------------------------
// 2) fused length-regulate + embedding lookup:  out = x_reg + pemb + eemb
// ---------------------------------------------------------------------------
__global__ __launch_bounds__(256)
void gather_kernel(const float* __restrict__ x, const float* __restrict__ xf,
                   const int* __restrict__ idx, const unsigned char* __restrict__ valid,
                   const float* __restrict__ ptgt, const float* __restrict__ etgt,
                   const float* __restrict__ pemb, const float* __restrict__ eemb,
                   const float* __restrict__ pbins, const float* __restrict__ ebins,
                   float* __restrict__ out, _Float16* __restrict__ xfh)
{
  const int f = blockIdx.x, b = blockIdx.y, ch = threadIdx.x;
  const size_t bf = (size_t)b * MAXLEN + f;
  const int id = idx[bf];
  const bool vld = valid[bf] != 0;
  const float pv = ptgt[bf], ev = etgt[bf];
  int pl = 0, ph = NB - 1;                 // searchsorted(bins, v, 'left')
  while (pl < ph) { int m = (pl + ph) >> 1; if (pbins[m] < pv) pl = m + 1; else ph = m; }
  int el = 0, eh = NB - 1;
  while (el < eh) { int m = (el + eh) >> 1; if (ebins[m] < ev) el = m + 1; else eh = m; }
  const size_t src = ((size_t)b * TSRC + id) * HDIM + ch;
  float xr  = vld ? x[src]  : 0.f;
  float xfr = vld ? xf[src] : 0.f;
  out[bf * HDIM + ch] = xr + pemb[(size_t)pl * HDIM + ch] + eemb[(size_t)el * HDIM + ch];
  xfh[bf * HDIM + ch] = (_Float16)xfr;
}

// ---------------------------------------------------------------------------
// 3a) generic f32 -> f16 conversion
// ---------------------------------------------------------------------------
__global__ void f32_to_f16_kernel(const float* __restrict__ in, _Float16* __restrict__ o, int n)
{
  for (int i = blockIdx.x * blockDim.x + threadIdx.x; i < n; i += gridDim.x * blockDim.x)
    o[i] = (_Float16)in[i];
}

// 3b) conv weight prep: f32 [768][256] (k,cin,cout) -> f16 transposed [256][768]
__global__ void wprep_kernel(const float* __restrict__ in, _Float16* __restrict__ o)
{
  int i = blockIdx.x * blockDim.x + threadIdx.x;      // over 768*256
  if (i < KIM * HDIM) {
    int kc = i >> 8, n = i & 255;
    o[(size_t)n * KIM + kc] = (_Float16)in[i];
  }
}

// ---------------------------------------------------------------------------
// 4) conv1d(K=3, 256->256) as WMMA GEMM (K_im2col = 768) + bias + ReLU + LN.
//    A tile staged by the Tensor Data Mover (TENSORcnt); weight chunks by
//    global_load_async_to_lds_b128 (ASYNCcnt), software-pipelined so the
//    async fill of chunk c+1 overlaps the 8 WMMAs of chunk c.
// ---------------------------------------------------------------------------
template<int MODE>
__global__ __launch_bounds__(256)
void conv_ln_kernel(const _Float16* __restrict__ inp,   // [B][T][256] f16
                    const _Float16* __restrict__ wT,    // [256][768]  f16 (n-major)
                    const float* __restrict__ bias,     // [256]
                    const float* __restrict__ gamma,    // [256]
                    const float* __restrict__ beta,     // [256]
                    const float* __restrict__ wl,       // [256]  (MODE 1)
                    const float* __restrict__ blin,     // [1]    (MODE 1)
                    const unsigned char* __restrict__ mask, int mask_is_valid,
                    _Float16* __restrict__ outh,        // MODE 0
                    float* __restrict__ outp,           // MODE 1
                    int T)
{
  __shared__ union SMem {
    struct { _Float16 A[66 * 264]; _Float16 B[256 * 40]; } ab;  // gemm phase
    _Float16 Y[64 * 264];                                       // store phase
  } u;
  __shared__ float sR1[4][16][2];
  __shared__ float sR2[4][16][2];

  _Float16* sA = u.ab.A;
  _Float16* sB = u.ab.B;
  const unsigned ldsA = (unsigned)(uintptr_t)sA;
  const unsigned ldsB = (unsigned)(uintptr_t)sB;

  const int tid  = threadIdx.x;
  const int b    = blockIdx.y;
  const int t0   = blockIdx.x * 64;
  const int wave = tid >> 5;
  const int lane = tid & 31;
  const int wm   = wave >> 1;      // 0..3 : 16-row slice
  const int wn   = wave & 1;       // 0..1 : 128-col slice
  const int ml   = lane & 15;
  const int lh   = lane >> 4;

  // async 2D fill of the weight chunk: sB[n][k], stride 40 halves (80B rows)
  auto fillB = [&](int kc0) {
    #pragma unroll
    for (int j = 0; j < 4; ++j) {
      int e = tid + j * 256;             // 1024 x 16B segments
      int n = e >> 2, seg = e & 3;
      ASYNC_CP16(ldsB + (unsigned)(n * 80 + seg * 16),
                 (const char*)(wT + (size_t)n * KIM + kc0) + seg * 16);
    }
  };

  // ---- stage A rows [t0-1, t0+64] into LDS ----
  const _Float16* xb = inp + (size_t)b * T * HDIM;
#if __has_builtin(__builtin_amdgcn_tensor_load_to_lds)
  {
    const int gr0   = (t0 > 0) ? (t0 - 1) : 0;       // first in-tensor halo row
    const int nrows = (t0 > 0) ? 66 : 65;            // trailing OOB rows auto-zeroed
    if (wave == 0)
      tdm_load_rows(ldsA + (t0 > 0 ? 0u : 528u),
                    xb + (size_t)gr0 * HDIM, nrows, T - gr0);
    if (t0 == 0)
      for (int i = tid; i < 32; i += 256)            // zero leading halo row
        *((uint4*)((char*)sA + i * 16)) = make_uint4(0u,0u,0u,0u);
  }
#else
  for (int i = tid; i < 66 * 32; i += 256) {         // async fallback
    int r = i >> 5, q = i & 31;
    int gr = t0 + r - 1;
    unsigned dst = ldsA + (unsigned)(r * 528 + q * 16);
    if (gr >= 0 && gr < T) {
      ASYNC_CP16(dst, (const char*)(xb + (size_t)gr * HDIM) + q * 16);
    } else {
      *((uint4*)((char*)sA + (size_t)r * 528 + q * 16)) = make_uint4(0u,0u,0u,0u);
    }
  }
#endif
  fillB(0);
  WAIT_ASYNC();
#if __has_builtin(__builtin_amdgcn_tensor_load_to_lds)
  if (wave == 0) __builtin_amdgcn_s_wait_tensorcnt(0);
#endif
  __syncthreads();

  v8f acc[8] = {};

  for (int chunk = 0; chunk < 24; ++chunk) {
    const int kc0 = chunk * 32;
    const int kk = kc0 >> 8;           // conv tap 0..2
    const int cb = kc0 & 255;          // input-channel base

    // A fragment (16x32 f16): lanes 0-15 -> K 0..7 & 16..23, lanes 16-31 -> K 8..15 & 24..31
    v16h a = {};
    {
      const int row = wm * 16 + ml + kk;              // = (t - t0 + 1) + (kk - 1)
      const char* base = (const char*)sA + (size_t)row * 528;
      const int kb = lh * 8;
      #pragma unroll
      for (int v = 0; v < 8; ++v) {
        int K = (v < 4) ? (kb + 2 * v) : (16 + kb + 2 * (v - 4));
        h2 p = *(const h2*)(base + (cb + K) * 2);
        a[2 * v] = p.x; a[2 * v + 1] = p.y;
      }
    }
    // B fragments (32x16 each): lane n = col, lanes 0-15 K 0..7 & 16..23, lanes 16-31 K 8..15 & 24..31
    v16h bf[8];
    #pragma unroll
    for (int s = 0; s < 8; ++s) {
      v16h t = {};
      const int n = wn * 128 + s * 16 + ml;
      const char* bbase = (const char*)sB + (size_t)n * 80;
      const int kb2 = lh * 8;
      #pragma unroll
      for (int v = 0; v < 8; ++v) {
        int K = (v < 4) ? (kb2 + 2 * v) : (16 + kb2 + 2 * (v - 4));
        h2 p = *(const h2*)(bbase + K * 2);
        t[2 * v] = p.x; t[2 * v + 1] = p.y;
      }
      bf[s] = t;
    }
    // all fragments in registers; barrier drains DScnt -> sB reusable
    __syncthreads();
    if (chunk + 1 < 24) {
      fillB(kc0 + 32);                               // async overlaps WMMAs below
      if (chunk + 2 < 24)
        __builtin_prefetch(wT + (size_t)tid * KIM + kc0 + 64, 0, 1);
      // pin: force the WMMAs (all consume `a`) to schedule AFTER the async
      // issues above; WAIT_ASYNC_DEP below keeps them BEFORE the wait.
      asm volatile("" : "+v"(a));
    }
    #pragma unroll
    for (int s = 0; s < 8; ++s)
      acc[s] = __builtin_amdgcn_wmma_f32_16x16x32_f16(
                 false, a, false, bf[s], (short)0, acc[s], false, false);
    if (chunk + 1 < 24) {
      WAIT_ASYNC_DEP(acc[0][0], acc[1][0], acc[2][0], acc[3][0],
                     acc[4][0], acc[5][0], acc[6][0], acc[7][0]);
      __syncthreads();
    }
  }
  __syncthreads();

  // ---- epilogue: bias + ReLU + LayerNorm (channels) in registers ----
  float bi[8], gg[8], be_[8], wl8[8];
  #pragma unroll
  for (int s = 0; s < 8; ++s) {
    int n = wn * 128 + s * 16 + ml;
    bi[s] = bias[n]; gg[s] = gamma[n]; be_[s] = beta[n];
    wl8[s] = (MODE == 1) ? wl[n] : 0.f;
  }
  float val[8][8], s1[8], s2[8];
  #pragma unroll
  for (int v = 0; v < 8; ++v) { s1[v] = 0.f; s2[v] = 0.f; }
  #pragma unroll
  for (int s = 0; s < 8; ++s)
    #pragma unroll
    for (int v = 0; v < 8; ++v) {
      float xv = acc[s][v] + bi[s];
      xv = xv > 0.f ? xv : 0.f;
      val[s][v] = xv; s1[v] += xv; s2[v] += xv * xv;
    }
  #pragma unroll
  for (int m = 1; m < 16; m <<= 1)
    #pragma unroll
    for (int v = 0; v < 8; ++v) {
      s1[v] += __shfl_xor(s1[v], m, 32);
      s2[v] += __shfl_xor(s2[v], m, 32);
    }
  if (ml == 0) {
    #pragma unroll
    for (int v = 0; v < 8; ++v) {
      sR1[wm][v + 8 * lh][wn] = s1[v];
      sR2[wm][v + 8 * lh][wn] = s2[v];
    }
  }
  __syncthreads();

  float pr[8];
  #pragma unroll
  for (int v = 0; v < 8; ++v) {
    const int row = v + 8 * lh;                       // 0..15 within wm slice
    float tot = sR1[wm][row][0] + sR1[wm][row][1];
    float tsq = sR2[wm][row][0] + sR2[wm][row][1];
    float mu  = tot * (1.f / 256.f);
    float var = tsq * (1.f / 256.f) - mu * mu;
    float rs  = rsqrtf(var + 1e-5f);
    float p = 0.f;
    #pragma unroll
    for (int s = 0; s < 8; ++s) {
      float yn = (val[s][v] - mu) * rs * gg[s] + be_[s];
      if (MODE == 0) {
        // stage to LDS (A/B buffers are dead) for coalesced b128 stores
        int n = wn * 128 + s * 16 + ml;
        u.Y[(size_t)(wm * 16 + row) * 264 + n] = (_Float16)yn;
      } else {
        p += yn * wl8[s];
      }
    }
    pr[v] = p;
  }

  if (MODE == 0) {
    __syncthreads();
    for (int i = tid; i < 64 * 32; i += 256) {
      int r = i >> 5, q = i & 31;
      uint4 v = *((const uint4*)((const char*)u.Y + (size_t)r * 528 + q * 16));
      ((uint4*)(outh + ((size_t)b * T + t0 + r) * HDIM))[q] = v;
    }
  } else {
    #pragma unroll
    for (int m = 1; m < 16; m <<= 1)
      #pragma unroll
      for (int v = 0; v < 8; ++v) pr[v] += __shfl_xor(pr[v], m, 32);
    __syncthreads();
    if (ml == 0)
      #pragma unroll
      for (int v = 0; v < 8; ++v) sR1[wm][v + 8 * lh][wn] = pr[v];
    __syncthreads();
    if (tid < 64) {
      const int r = tid;
      float p = sR1[r >> 4][r & 15][0] + sR1[r >> 4][r & 15][1] + blin[0];
      const int t = t0 + r;
      unsigned char mk = mask[(size_t)b * T + t];
      bool masked = mask_is_valid ? (mk == 0) : (mk != 0);
      outp[(size_t)b * T + t] = masked ? 0.f : p;
    }
  }
}

// ---------------------------------------------------------------------------
extern "C" void kernel_launch(void* const* d_in, const int* in_sizes, int n_in,
                              void* d_out, int out_size, void* d_ws, size_t ws_size,
                              hipStream_t stream)
{
  (void)in_sizes; (void)n_in; (void)out_size; (void)ws_size;

  // ---- inputs (setup_inputs order; params dicts flatten in insertion order) ----
  const float* x    = (const float*)d_in[0];
  const float* xf   = (const float*)d_in[1];
  const unsigned char* smask = (const unsigned char*)d_in[2];
  const int*   dur  = (const int*)d_in[3];
  const float* ptgt = (const float*)d_in[4];
  const float* etgt = (const float*)d_in[5];
  const float* P[3][10];   // [dur|pitch|energy][w1,b1,g1,be1,w2,b2,g2,be2,wl,bl]
  for (int p = 0; p < 3; ++p)
    for (int q = 0; q < 10; ++q)
      P[p][q] = (const float*)d_in[7 + p * 10 + q];
  const float* pemb  = (const float*)d_in[37];
  const float* eemb  = (const float*)d_in[38];
  const float* pbins = (const float*)d_in[39];
  const float* ebins = (const float*)d_in[40];

  // ---- output layout (tuple concatenated flat) ----
  float* out       = (float*)d_out;
  float* o_out     = out;                              // 64*2048*256
  float* o_logdur  = o_out    + (size_t)BATCH * MAXLEN * HDIM;
  float* o_durr    = o_logdur + (size_t)BATCH * TSRC;
  float* o_pitch   = o_durr   + (size_t)BATCH * TSRC;
  float* o_energy  = o_pitch  + (size_t)BATCH * MAXLEN;
  float* o_mellen  = o_energy + (size_t)BATCH * MAXLEN;
  float* o_melmask = o_mellen + (size_t)BATCH;

  // ---- workspace layout ----
  char* w = (char*)d_ws;
  _Float16* xfh_reg = (_Float16*)w; w += (size_t)BATCH * MAXLEN * HDIM * 2;   // 67 MB
  _Float16* xfeat_h = (_Float16*)w; w += (size_t)BATCH * TSRC * HDIM * 2;     // 17 MB
  _Float16* y1      = (_Float16*)w; w += (size_t)BATCH * MAXLEN * HDIM * 2;   // 67 MB
  _Float16* wh[6];
  for (int i = 0; i < 6; ++i) { wh[i] = (_Float16*)w; w += (size_t)KIM * HDIM * 2; }
  int* idx = (int*)w;              w += (size_t)BATCH * MAXLEN * 4;
  unsigned char* valid = (unsigned char*)w;

  // 1) duration cumsum -> gather idx / masks / lengths
  idx_kernel<<<BATCH, TSRC, 0, stream>>>(dur, idx, valid, o_durr, o_mellen, o_melmask);

  // 2) f16 conversions: x_features; conv weights converted + transposed to [256][768]
  f32_to_f16_kernel<<<2048, 256, 0, stream>>>(xf, xfeat_h, BATCH * TSRC * HDIM);
  for (int p = 0; p < 3; ++p) {
    wprep_kernel<<<KIM, 256, 0, stream>>>(P[p][0], wh[p * 2 + 0]);
    wprep_kernel<<<KIM, 256, 0, stream>>>(P[p][4], wh[p * 2 + 1]);
  }

  // 3) length-regulate + embedding add
  gather_kernel<<<dim3(MAXLEN, BATCH), 256, 0, stream>>>(
      x, xf, idx, valid, ptgt, etgt, pemb, eemb, pbins, ebins, o_out, xfh_reg);

  // 4) duration predictor (T = 512) on x_features
  conv_ln_kernel<0><<<dim3(TSRC / 64, BATCH), 256, 0, stream>>>(
      xfeat_h, wh[0], P[0][1], P[0][2], P[0][3], P[0][8], P[0][9],
      smask, 0, y1, nullptr, TSRC);
  conv_ln_kernel<1><<<dim3(TSRC / 64, BATCH), 256, 0, stream>>>(
      y1, wh[1], P[0][5], P[0][6], P[0][7], P[0][8], P[0][9],
      smask, 0, nullptr, o_logdur, TSRC);

  // 5) pitch predictor (T = 2048) on regulated features (mask = !valid)
  conv_ln_kernel<0><<<dim3(MAXLEN / 64, BATCH), 256, 0, stream>>>(
      xfh_reg, wh[2], P[1][1], P[1][2], P[1][3], P[1][8], P[1][9],
      valid, 1, y1, nullptr, MAXLEN);
  conv_ln_kernel<1><<<dim3(MAXLEN / 64, BATCH), 256, 0, stream>>>(
      y1, wh[3], P[1][5], P[1][6], P[1][7], P[1][8], P[1][9],
      valid, 1, nullptr, o_pitch, MAXLEN);

  // 6) energy predictor (T = 2048)
  conv_ln_kernel<0><<<dim3(MAXLEN / 64, BATCH), 256, 0, stream>>>(
      xfh_reg, wh[4], P[2][1], P[2][2], P[2][3], P[2][8], P[2][9],
      valid, 1, y1, nullptr, MAXLEN);
  conv_ln_kernel<1><<<dim3(MAXLEN / 64, BATCH), 256, 0, stream>>>(
      y1, wh[5], P[2][5], P[2][6], P[2][7], P[2][8], P[2][9],
      valid, 1, nullptr, o_energy, MAXLEN);
}
